// FeaturePropagation_50654844289751
// MI455X (gfx1250) — matrix operand order repaired and checked by reading, not compiled, not run on hardware
//
#include <hip/hip_runtime.h>
#include <hip/hip_bf16.h>
#include <stdint.h>

typedef __attribute__((ext_vector_type(16))) _Float16 v16h;
typedef __attribute__((ext_vector_type(8)))  float    v8f;
typedef int v4i_gcc __attribute__((vector_size(16)));

#define B_   8
#define N_   8192
#define S_   2048
#define C1_  128
#define C2_  256
#define CIN_ 384
#define H_   256
#define EPS_LN 1e-5f

#define AS1 __attribute__((address_space(1)))
#define AS3 __attribute__((address_space(3)))

// ---- gfx1250 async global->LDS path (guarded; falls back to load+ds_store) ----
#if __has_builtin(__builtin_amdgcn_global_load_async_to_lds_b128)
#define HAVE_ASYNC_LDS 1
__device__ __forceinline__ void async_copy_b128(const void* g, void* l) {
    __builtin_amdgcn_global_load_async_to_lds_b128(
        (AS1 v4i_gcc*)g, (AS3 v4i_gcc*)l, 0, 0);
}
#else
#define HAVE_ASYNC_LDS 0
#endif

__device__ __forceinline__ void wait_async_zero() {
#if __has_builtin(__builtin_amdgcn_s_wait_asynccnt)
    __builtin_amdgcn_s_wait_asynccnt(0);
#else
    asm volatile("s_wait_asynccnt 0x0" ::: "memory");
#endif
}

// ---------------------------------------------------------------------------
// 1) big = 3 * max|coor2| over the whole [B,3,S] tensor
// ---------------------------------------------------------------------------
__global__ void k_maxabs(const float* __restrict__ c2, float* __restrict__ big) {
    __shared__ float red[256];
    float m = 0.f;
    const int total = B_ * 3 * S_;
    for (int i = threadIdx.x; i < total; i += 256) m = fmaxf(m, fabsf(c2[i]));
    red[threadIdx.x] = m;
    __syncthreads();
    for (int s = 128; s > 0; s >>= 1) {
        if (threadIdx.x < s) red[threadIdx.x] = fmaxf(red[threadIdx.x], red[threadIdx.x + s]);
        __syncthreads();
    }
    if (threadIdx.x == 0) big[0] = red[0] * 3.0f;
}

// ---------------------------------------------------------------------------
// 2) 3-NN over S=2048 sources (staged in LDS) + inverse-distance weights
// ---------------------------------------------------------------------------
__global__ void k_knn(const float* __restrict__ c1, const float* __restrict__ c2,
                      const uint8_t* __restrict__ pad, const float* __restrict__ bigp,
                      int* __restrict__ idx, float* __restrict__ wgt) {
    __shared__ float sx[S_], sy[S_], sz[S_];
    const int b = blockIdx.y;
    const float big = bigp[0];
    for (int s = threadIdx.x; s < S_; s += 256) {
        const bool p = pad[(size_t)b * S_ + s] != 0;
        float x = c2[((size_t)b * 3 + 0) * S_ + s];
        float y = c2[((size_t)b * 3 + 1) * S_ + s];
        float z = c2[((size_t)b * 3 + 2) * S_ + s];
        sx[s] = p ? big : x;
        sy[s] = p ? big : y;
        sz[s] = p ? big : z;
    }
    __syncthreads();

    const int n = blockIdx.x * 256 + threadIdx.x;
    const float qx = c1[((size_t)b * 3 + 0) * N_ + n];
    const float qy = c1[((size_t)b * 3 + 1) * N_ + n];
    const float qz = c1[((size_t)b * 3 + 2) * N_ + n];

    float d0 = 3.0e38f, d1 = 3.0e38f, d2 = 3.0e38f;
    int   i0 = 0, i1 = 0, i2 = 0;
#pragma unroll 4
    for (int s = 0; s < S_; ++s) {
        const float dx = qx - sx[s], dy = qy - sy[s], dz = qz - sz[s];
        const float d = dx * dx + dy * dy + dz * dz;
        if (d < d2) {
            if (d < d1) {
                if (d < d0) { d2 = d1; i2 = i1; d1 = d0; i1 = i0; d0 = d; i0 = s; }
                else        { d2 = d1; i2 = i1; d1 = d;  i1 = s; }
            } else          { d2 = d;  i2 = s; }
        }
    }
    const float r0 = 1.f / fmaxf(d0, 1e-8f);
    const float r1 = 1.f / fmaxf(d1, 1e-8f);
    const float r2 = 1.f / fmaxf(d2, 1e-8f);
    const float rs = 1.f / (r0 + r1 + r2);
    const size_t base = ((size_t)b * N_ + n) * 3;
    idx[base + 0] = i0; idx[base + 1] = i1; idx[base + 2] = i2;
    wgt[base + 0] = r0 * rs; wgt[base + 1] = r1 * rs; wgt[base + 2] = r2 * rs;
}

// ---------------------------------------------------------------------------
// 3) fp32 -> f16 weight conversion
// ---------------------------------------------------------------------------
__global__ void k_cvt(const float* __restrict__ W1, const float* __restrict__ W2,
                      _Float16* __restrict__ W1h, _Float16* __restrict__ W2h) {
    const int i = blockIdx.x * 256 + threadIdx.x;
    if (i < H_ * CIN_) W1h[i] = (_Float16)W1[i];
    if (i < H_ * H_)   W2h[i] = (_Float16)W2[i];
}

// ---------------------------------------------------------------------------
// 4) Build X[B][N][384] f16 = concat(fea1, IDW-interp(fea2)) point-major
// ---------------------------------------------------------------------------
__global__ void k_buildx(const float* __restrict__ fea1, const float* __restrict__ fea2,
                         const int* __restrict__ idx, const float* __restrict__ wgt,
                         _Float16* __restrict__ X) {
    const int PITCH = 392;                 // padded row pitch (halves), 16B aligned
    __shared__ _Float16 tile[64 * PITCH];  // ~49 KB
    const int b  = blockIdx.y;
    const int n0 = blockIdx.x * 64;

#pragma unroll
    for (int it = 0; it < 32; ++it) {
        const int e = it * 256 + threadIdx.x;
        const int n = e & 63, c = e >> 6;
        tile[n * PITCH + c] = (_Float16)fea1[((size_t)b * C1_ + c) * N_ + (n0 + n)];
    }
    const int c = threadIdx.x;
    const float* f2c = fea2 + ((size_t)b * C2_ + c) * S_;
    for (int n = 0; n < 64; ++n) {
        const size_t ib = ((size_t)b * N_ + n0 + n) * 3;
        const int   j0 = idx[ib], j1 = idx[ib + 1], j2 = idx[ib + 2];
        const float w0 = wgt[ib], w1 = wgt[ib + 1], w2 = wgt[ib + 2];
        const float v  = w0 * f2c[j0] + w1 * f2c[j1] + w2 * f2c[j2];
        tile[n * PITCH + C1_ + c] = (_Float16)v;
    }
    __syncthreads();
#pragma unroll
    for (int it = 0; it < 12; ++it) {
        const int chunk = it * 256 + threadIdx.x;  // 64*48 = 3072 uint4 chunks
        const int n = chunk / 48, off = chunk % 48;
        const uint4 v = *(const uint4*)&tile[n * PITCH + off * 8];
        *(uint4*)&X[(((size_t)b * N_ + n0 + n) * CIN_) + off * 8] = v;
    }
}

// ---------------------------------------------------------------------------
// 5) Fused GEMM (WMMA f16->f32) + bias + channel-LayerNorm + ReLU
//    Double-buffered LDS tiles; async global->LDS when available.
//    A = W[H][KD] f16, B = Xsrc[B][N][KD] f16 (point-major)
//    EPI==0: store f16 [B][N][H]; EPI==1: store f32 [B][H][N]
// ---------------------------------------------------------------------------
template <int KD, int EPI>
__global__ void k_gemm_ln(const _Float16* __restrict__ A,
                          const _Float16* __restrict__ Xsrc,
                          const float* __restrict__ bias,
                          const float* __restrict__ gamma,
                          const float* __restrict__ beta,
                          _Float16* __restrict__ Yh,
                          float* __restrict__ Yf) {
    constexpr int AP = 40, BP = 40;        // padded pitches (halves)
    constexpr int ASZ = 256 * AP, BSZ = 64 * BP;
    constexpr int NK = KD / 32;
    __shared__ _Float16 As[2 * ASZ];       // 2 x 20 KB
    __shared__ _Float16 Bs[2 * BSZ];       // 2 x 5 KB
    __shared__ float ssum[64], ssq[64];

    const int b    = blockIdx.y;
    const int n0   = blockIdx.x * 64;
    const int tid  = threadIdx.x;
    const int lane = tid & 31;
    const int wave = tid >> 5;
    const int half16 = (lane >> 4) & 1;    // K-half select per 16-bit WMMA layout
    const int lr     = lane & 15;          // row (A) / col (B) within tile

    const int bn = tid >> 2, bko = (tid & 3) * 8;  // B-tile copy assignment

    auto load_tiles = [&](int kt, int p) {
        const _Float16* ga = A + (size_t)tid * KD + kt * 32;
        _Float16*       la = As + p * ASZ + tid * AP;
        const _Float16* gb = Xsrc + ((size_t)b * N_ + n0 + bn) * KD + kt * 32 + bko;
        _Float16*       lb = Bs + p * BSZ + bn * BP + bko;
#if HAVE_ASYNC_LDS
        async_copy_b128(ga +  0, la +  0);
        async_copy_b128(ga +  8, la +  8);
        async_copy_b128(ga + 16, la + 16);
        async_copy_b128(ga + 24, la + 24);
        async_copy_b128(gb, lb);
#else
        const uint4* s4 = (const uint4*)ga;
        uint4*       d4 = (uint4*)la;
        d4[0] = s4[0]; d4[1] = s4[1]; d4[2] = s4[2]; d4[3] = s4[3];
        *(uint4*)lb = *(const uint4*)gb;
#endif
    };

    const v8f vzero = {0.f, 0.f, 0.f, 0.f, 0.f, 0.f, 0.f, 0.f};
    v8f acc[2][4];
#pragma unroll
    for (int i = 0; i < 2; ++i)
#pragma unroll
        for (int j = 0; j < 4; ++j) acc[i][j] = vzero;

    load_tiles(0, 0);
#if HAVE_ASYNC_LDS
    wait_async_zero();
#endif
    __syncthreads();

    for (int kt = 0; kt < NK; ++kt) {
        const int p = kt & 1;
        if (kt + 1 < NK) load_tiles(kt + 1, p ^ 1);

        const _Float16* Ab = As + p * ASZ;
        const _Float16* Bb = Bs + p * BSZ;
#pragma unroll
        for (int j = 0; j < 4; ++j) {
            // B fragment: lane holds col n=lr, lane<16 K{0..7,16..23}, lane>=16 K{8..15,24..31}
            union { uint4 q[2]; v16h h; } bf;
            const _Float16* bp = Bb + (j * 16 + lr) * BP + half16 * 8;
            bf.q[0] = *(const uint4*)(bp);
            bf.q[1] = *(const uint4*)(bp + 16);
#pragma unroll
            for (int i = 0; i < 2; ++i) {
                // A fragment: lane<16 K{0..7,16..23}, lane>=16 K{8..15,24..31}
                union { uint4 q[2]; v16h h; } af;
                const _Float16* ap = Ab + (wave * 32 + i * 16 + lr) * AP + half16 * 8;
                af.q[0] = *(const uint4*)(ap);
                af.q[1] = *(const uint4*)(ap + 16);
                acc[i][j] = __builtin_amdgcn_wmma_f32_16x16x32_f16(
                    false, af.h, false, bf.h, (short)0, acc[i][j], false, false);
            }
        }
#if HAVE_ASYNC_LDS
        wait_async_zero();
#endif
        __syncthreads();
    }

    // ---- epilogue: bias + LayerNorm(channel) + ReLU ----
    if (tid < 64) { ssum[tid] = 0.f; ssq[tid] = 0.f; }
    __syncthreads();
#pragma unroll
    for (int i = 0; i < 2; ++i) {
        const int m0 = wave * 32 + i * 16 + half16 * 8;  // 8 contiguous channels
        float bv[8];
#pragma unroll
        for (int r = 0; r < 8; ++r) bv[r] = bias[m0 + r];
#pragma unroll
        for (int j = 0; j < 4; ++j) {
            float ls = 0.f, lq = 0.f;
#pragma unroll
            for (int r = 0; r < 8; ++r) {
                const float v = acc[i][j][r] + bv[r];
                acc[i][j][r] = v;
                ls += v; lq += v * v;
            }
            const int nl = j * 16 + lr;
            atomicAdd(&ssum[nl], ls);   // ds_add_f32
            atomicAdd(&ssq[nl], lq);
        }
    }
    __syncthreads();
#pragma unroll
    for (int i = 0; i < 2; ++i) {
        const int m0 = wave * 32 + i * 16 + half16 * 8;
        float gv[8], ev[8];
#pragma unroll
        for (int r = 0; r < 8; ++r) { gv[r] = gamma[m0 + r]; ev[r] = beta[m0 + r]; }
#pragma unroll
        for (int j = 0; j < 4; ++j) {
            const int nl   = j * 16 + lr;
            const float mu = ssum[nl] * (1.0f / H_);
            const float va = ssq[nl] * (1.0f / H_) - mu * mu;
            const float rs = rsqrtf(va + EPS_LN);
            if (EPI == 0) {
                union { uint4 q; _Float16 h[8]; } o;
#pragma unroll
                for (int r = 0; r < 8; ++r) {
                    const float v = gv[r] * (acc[i][j][r] - mu) * rs + ev[r];
                    o.h[r] = (_Float16)fmaxf(v, 0.f);
                }
                *(uint4*)(Yh + ((size_t)b * N_ + n0 + nl) * H_ + m0) = o.q;
            } else {
#pragma unroll
                for (int r = 0; r < 8; ++r) {
                    const float v = gv[r] * (acc[i][j][r] - mu) * rs + ev[r];
                    Yf[((size_t)b * H_ + m0 + r) * N_ + n0 + nl] = fmaxf(v, 0.f);
                }
            }
        }
    }
}

// ---------------------------------------------------------------------------
extern "C" void kernel_launch(void* const* d_in, const int* in_sizes, int n_in,
                              void* d_out, int out_size, void* d_ws, size_t ws_size,
                              hipStream_t stream) {
    const float*   coor1 = (const float*)d_in[0];
    const float*   coor2 = (const float*)d_in[1];
    const float*   fea1  = (const float*)d_in[2];
    const float*   fea2  = (const float*)d_in[3];
    const uint8_t* pad   = (const uint8_t*)d_in[4];
    const float* W1  = (const float*)d_in[5];
    const float* b1  = (const float*)d_in[6];
    const float* g1  = (const float*)d_in[7];
    const float* be1 = (const float*)d_in[8];
    const float* W2  = (const float*)d_in[9];
    const float* b2  = (const float*)d_in[10];
    const float* g2  = (const float*)d_in[11];
    const float* be2 = (const float*)d_in[12];
    float* out = (float*)d_out;

    char* ws = (char*)d_ws;
    size_t off = 0;
    float* big = (float*)(ws + off);      off += 256;
    int*   idx = (int*)(ws + off);        off += (size_t)B_ * N_ * 3 * sizeof(int);
    float* wgt = (float*)(ws + off);      off += (size_t)B_ * N_ * 3 * sizeof(float);
    _Float16* W1h = (_Float16*)(ws + off); off += (size_t)H_ * CIN_ * 2;
    _Float16* W2h = (_Float16*)(ws + off); off += (size_t)H_ * H_ * 2;
    _Float16* Xh  = (_Float16*)(ws + off); off += (size_t)B_ * N_ * CIN_ * 2;
    _Float16* Y1h = (_Float16*)(ws + off); off += (size_t)B_ * N_ * H_ * 2;
    (void)ws_size; (void)in_sizes; (void)n_in; (void)out_size;

    k_maxabs<<<1, 256, 0, stream>>>(coor2, big);
    k_knn<<<dim3(N_ / 256, B_), 256, 0, stream>>>(coor1, coor2, pad, big, idx, wgt);
    k_cvt<<<(H_ * CIN_ + 255) / 256, 256, 0, stream>>>(W1, W2, W1h, W2h);
    k_buildx<<<dim3(N_ / 64, B_), 256, 0, stream>>>(fea1, fea2, idx, wgt, Xh);
    k_gemm_ln<CIN_, 0><<<dim3(N_ / 64, B_), 256, 0, stream>>>(W1h, Xh, b1, g1, be1, Y1h, nullptr);
    k_gemm_ln<H_, 1><<<dim3(N_ / 64, B_), 256, 0, stream>>>(W2h, Y1h, b2, g2, be2, nullptr, out);
}
